// AugustAttention_56788057587864
// MI455X (gfx1250) — compile-verified, matrix operand order checked by reading, not run on hardware
//
#include <hip/hip_runtime.h>
#include <stdint.h>

// ---------------------------------------------------------------------------
// CDNA5 (gfx1250) attention: projections + RoPE + causal softmax attention.
// All matmuls via v_wmma_f32_16x16x32_bf16 (wave32), double-buffered LDS
// staging, async global->LDS copies (ASYNCcnt) where operands are bf16.
// ---------------------------------------------------------------------------

typedef __attribute__((ext_vector_type(16))) __bf16 v16bf;
typedef __attribute__((ext_vector_type(8)))  float  v8f;
typedef __attribute__((ext_vector_type(4)))  int    v4i;

__device__ __forceinline__ unsigned short f2bf(float f) {
  union { float f; uint32_t u; } x; x.f = f;
  uint32_t r = (x.u + 0x7FFFu + ((x.u >> 16) & 1u)) >> 16;
  return (unsigned short)r;
}
__device__ __forceinline__ float bf2f(unsigned short h) {
  union { uint32_t u; float f; } x; x.u = ((uint32_t)h) << 16;
  return x.f;
}
__device__ __forceinline__ uint32_t pk2bf(float lo, float hi) {
  return (uint32_t)f2bf(lo) | ((uint32_t)f2bf(hi) << 16);
}

// ---- async global->LDS (gfx1250 GLOBAL_LOAD_ASYNC_TO_LDS_B128) -------------
#if defined(__gfx1250__) && __has_builtin(__builtin_amdgcn_global_load_async_to_lds_b128)
#define USE_ASYNC_LDS 1
#else
#define USE_ASYNC_LDS 0
#endif

__device__ __forceinline__ void async_copy16(const unsigned short* gsrc,
                                             unsigned short* ldst) {
#if USE_ASYNC_LDS
  // Param 0: int4 addrspace(1)* (global src), param 1: int4 addrspace(3)* (LDS
  // dst), then imm offset and cpol. Build AS pointers via integers: AS1 ==
  // flat global address; AS3 pointer is 32-bit, truncation of the flat shared
  // address yields the LDS offset (ISA: LDS aperture discards upper bits).
  __builtin_amdgcn_global_load_async_to_lds_b128(
      (__attribute__((address_space(1))) v4i*)(uintptr_t)gsrc,
      (__attribute__((address_space(3))) v4i*)(uintptr_t)ldst, 0, 0);
#else
  *(uint4*)ldst = *(const uint4*)gsrc;
#endif
}

__device__ __forceinline__ void async_fence() {
#if USE_ASYNC_LDS
#if __has_builtin(__builtin_amdgcn_s_wait_asynccnt)
  __builtin_amdgcn_s_wait_asynccnt(0);
#else
  asm volatile("s_wait_asynccnt 0x0" ::: "memory");
#endif
#endif
}

// Block tile: 128x128, K-step 32. 256 threads = 8 waves (wave32).
// Wave grid 4 (M) x 2 (N): each wave computes 32 rows x 64 cols = 2x4 WMMA tiles.
//
// AM: 0 = A is fp32 row-major (M x K), 1 = A is bf16 row-major (M x K)
// BM: 0 = B is fp32 (K x N), 1 = B is bf16 (K x N), 2 = B is bf16 (N x K)  [Q*K^T]
// EPI: 0 = store bf16
//      1 = scores: val*scale, causal mask (col>row -> -1e30), store bf16
//      2 = P*V: val / rowsum[row], store bf16, K-loop clipped at causal bound
//      3 = store fp32
template <int AM, int BM, int EPI>
__global__ __launch_bounds__(256) void wmma_gemm(
    const void* __restrict__ Ag, const void* __restrict__ Bg, void* __restrict__ Cg,
    int M, int N, int K, int lda, int ldb, int ldc,
    size_t sA, size_t sB, size_t sC,
    float scale, const float* __restrict__ rs, size_t sR) {
  const int z  = blockIdx.z;
  const int m0 = blockIdx.y * 128;
  const int n0 = blockIdx.x * 128;
  const int tid  = threadIdx.x;
  const int lane = tid & 31;
  const int wave = tid >> 5;
  const int wr   = wave & 3;  // wave row: 32 rows each
  const int wc   = wave >> 2; // wave col: 64 cols each

  const float*          A_f = (const float*)Ag          + (size_t)z * sA;
  const unsigned short* A_h = (const unsigned short*)Ag + (size_t)z * sA;
  const float*          B_f = (const float*)Bg          + (size_t)z * sB;
  const unsigned short* B_h = (const unsigned short*)Bg + (size_t)z * sB;
  float*                C_f = (float*)Cg                + (size_t)z * sC;
  unsigned short*       C_h = (unsigned short*)Cg       + (size_t)z * sC;
  const float*          rsz = (EPI == 2) ? rs + (size_t)z * sR : nullptr;

  // Fully-masked causal tile: fill with -1e30 and exit.
  if (EPI == 1 && n0 >= m0 + 128) {
    const unsigned short nv = f2bf(-1e30f);
    for (int i = 0; i < 64; ++i) {
      int e = tid + i * 256;
      int r = e >> 7, c = e & 127;
      C_h[(size_t)(m0 + r) * ldc + (n0 + c)] = nv;
    }
    return;
  }

  // Double-buffered tiles: A [m][k], B transposed [n][k], bf16.
  __shared__ __align__(16) unsigned short lA[2][128][32];
  __shared__ __align__(16) unsigned short lB[2][128][32];

  v8f acc[2][4];
  #pragma unroll
  for (int mi = 0; mi < 2; ++mi)
    #pragma unroll
    for (int ni = 0; ni < 4; ++ni)
      acc[mi][ni] = (v8f){0.f, 0.f, 0.f, 0.f, 0.f, 0.f, 0.f, 0.f};

  int kEnd = K;
  if (EPI == 2) { int kl = m0 + 128; kEnd = (kl < K) ? kl : K; }  // causal clip

  // Stage one 128x32 A tile + 128x32 (transposed) B tile into buffer `buf`.
  auto stage = [&](int k0, int buf) {
    // ---- A tile ----
    if (AM == 1) {
      #pragma unroll
      for (int i = 0; i < 2; ++i) {              // 512 x 16B chunks / 256 thr
        int e = tid + i * 256;
        int r = e >> 2, c = (e & 3) * 8;
        async_copy16(&A_h[(size_t)(m0 + r) * lda + (k0 + c)], &lA[buf][r][c]);
      }
    } else {
      #pragma unroll
      for (int i = 0; i < 4; ++i) {              // 1024 x float4 chunks
        int e = tid + i * 256;
        int r = e >> 3, c = (e & 7) * 4;
        float4 f = *(const float4*)&A_f[(size_t)(m0 + r) * lda + (k0 + c)];
        uint2 p; p.x = pk2bf(f.x, f.y); p.y = pk2bf(f.z, f.w);
        *(uint2*)&lA[buf][r][c] = p;
      }
    }
    // ---- B tile (stored [n][k]) ----
    if (BM == 2) {                               // physically N x K: pure copy
      #pragma unroll
      for (int i = 0; i < 2; ++i) {
        int e = tid + i * 256;
        int r = e >> 2, c = (e & 3) * 8;
        async_copy16(&B_h[(size_t)(n0 + r) * ldb + (k0 + c)], &lB[buf][r][c]);
      }
    } else if (BM == 1) {                        // bf16 K x N: transpose in LDS
      #pragma unroll
      for (int i = 0; i < 2; ++i) {              // 512 x (8 bf16) chunks
        int e = tid + i * 256;
        int r = e >> 4, c = (e & 15) * 8;        // r: K row, c: N col base
        union { uint4 q; unsigned short u[8]; } d;
        d.q = *(const uint4*)&B_h[(size_t)(k0 + r) * ldb + (n0 + c)];
        #pragma unroll
        for (int j = 0; j < 8; ++j) lB[buf][c + j][r] = d.u[j];
      }
    } else {                                     // fp32 K x N: convert+transpose
      #pragma unroll
      for (int i = 0; i < 4; ++i) {              // 1024 x float4 chunks
        int e = tid + i * 256;
        int r = e >> 5, c = (e & 31) * 4;
        float4 f = *(const float4*)&B_f[(size_t)(k0 + r) * ldb + (n0 + c)];
        lB[buf][c + 0][r] = f2bf(f.x);
        lB[buf][c + 1][r] = f2bf(f.y);
        lB[buf][c + 2][r] = f2bf(f.z);
        lB[buf][c + 3][r] = f2bf(f.w);
      }
    }
  };

  union Frag { v16bf v; uint4 q[2]; };
  const int koff = (lane >> 4) << 3;  // lanes 0-15: K 0-7/16-23; lanes 16-31: K 8-15/24-31

  stage(0, 0);

  const int steps = kEnd >> 5;
  for (int it = 0; it < steps; ++it) {
    const int cur = it & 1;
    async_fence();        // this wave's copies into buf `cur` are in LDS
    __syncthreads();      // everyone's copies landed; prior reads of 1-cur done

    if (it + 1 < steps) stage((it + 1) << 5, 1 - cur);

    // ---- fragments from buf `cur` ----
    Frag a[2], b[4];
    #pragma unroll
    for (int mi = 0; mi < 2; ++mi) {
      const unsigned short* pr = &lA[cur][wr * 32 + mi * 16 + (lane & 15)][0];
      a[mi].q[0] = *(const uint4*)(pr + koff);
      a[mi].q[1] = *(const uint4*)(pr + koff + 16);
    }
    #pragma unroll
    for (int ni = 0; ni < 4; ++ni) {
      const unsigned short* pr = &lB[cur][wc * 64 + ni * 16 + (lane & 15)][0];
      b[ni].q[0] = *(const uint4*)(pr + koff);
      b[ni].q[1] = *(const uint4*)(pr + koff + 16);
    }

    // ---- 8 WMMAs: D = A x B + C ----
    #pragma unroll
    for (int mi = 0; mi < 2; ++mi)
      #pragma unroll
      for (int ni = 0; ni < 4; ++ni)
        acc[mi][ni] = __builtin_amdgcn_wmma_f32_16x16x32_bf16(
            false, a[mi].v, false, b[ni].v, (short)0, acc[mi][ni], false, false);
  }

  // ---- epilogue ----
  const int cn = wc * 64 + (lane & 15);
  const int mh = (lane >> 4) << 3;  // +8 rows for upper half-wave
  #pragma unroll
  for (int mi = 0; mi < 2; ++mi) {
    #pragma unroll
    for (int ni = 0; ni < 4; ++ni) {
      int gn = n0 + cn + ni * 16;
      #pragma unroll
      for (int v = 0; v < 8; ++v) {
        int gm = m0 + wr * 32 + mi * 16 + mh + v;
        float val = acc[mi][ni][v];
        if (EPI == 0) {
          C_h[(size_t)gm * ldc + gn] = f2bf(val);
        } else if (EPI == 1) {
          val *= scale;
          if (gn > gm) val = -1e30f;
          C_h[(size_t)gm * ldc + gn] = f2bf(val);
        } else if (EPI == 2) {
          float d = rsz[gm];
          C_h[(size_t)gm * ldc + gn] = f2bf(val / d);
        } else {
          C_f[(size_t)gm * ldc + gn] = val;
        }
      }
    }
  }
}

// In-place RoPE on bf16 Q and K. One thread per (b, s, pair).
__global__ __launch_bounds__(256) void rope_kernel(
    unsigned short* __restrict__ Qb, unsigned short* __restrict__ Kb, int S, int D) {
  size_t gid  = (size_t)blockIdx.x * blockDim.x + threadIdx.x;
  int    half = D >> 1;
  int    p    = (int)(gid % half);
  size_t sd   = gid / half;          // b*S + s
  int    s    = (int)(sd % S);
  float inv = __expf(-((float)(2 * p) / (float)D) * 9.210340371976184f);
  float ang = (float)s * inv;
  float sn, cs;
  __sincosf(ang, &sn, &cs);
  size_t base = sd * (size_t)D + (size_t)(2 * p);
  float x1 = bf2f(Qb[base]), x2 = bf2f(Qb[base + 1]);
  Qb[base]     = f2bf(x1 * cs - x2 * sn);
  Qb[base + 1] = f2bf(x1 * sn + x2 * cs);
  x1 = bf2f(Kb[base]); x2 = bf2f(Kb[base + 1]);
  Kb[base]     = f2bf(x1 * cs - x2 * sn);
  Kb[base + 1] = f2bf(x1 * sn + x2 * cs);
}

// Row softmax over bf16 scores, in place. Row (S=4096) held in registers.
__global__ __launch_bounds__(256) void softmax_rows(
    unsigned short* __restrict__ P, float* __restrict__ rs, int S) {
  const int row = blockIdx.x;
  const int z   = blockIdx.y;
  unsigned short* p = P + (size_t)z * S * S + (size_t)row * S;
  const int t = threadIdx.x;

  float vals[16];
  #pragma unroll
  for (int i = 0; i < 16; ++i) vals[i] = bf2f(p[t + i * 256]);

  __shared__ float red[256];
  float m = -3.0e38f;
  #pragma unroll
  for (int i = 0; i < 16; ++i) m = fmaxf(m, vals[i]);
  red[t] = m;
  __syncthreads();
  for (int o = 128; o > 0; o >>= 1) {
    if (t < o) red[t] = fmaxf(red[t], red[t + o]);
    __syncthreads();
  }
  m = red[0];
  __syncthreads();

  float s = 0.f;
  #pragma unroll
  for (int i = 0; i < 16; ++i) { vals[i] = __expf(vals[i] - m); s += vals[i]; }
  red[t] = s;
  __syncthreads();
  for (int o = 128; o > 0; o >>= 1) {
    if (t < o) red[t] += red[t + o];
    __syncthreads();
  }
  s = red[0];

  #pragma unroll
  for (int i = 0; i < 16; ++i) p[t + i * 256] = f2bf(vals[i]);
  if (t == 0) rs[(size_t)z * S + row] = s;
}

extern "C" void kernel_launch(void* const* d_in, const int* in_sizes, int n_in,
                              void* d_out, int out_size, void* d_ws, size_t ws_size,
                              hipStream_t stream) {
  (void)in_sizes; (void)n_in; (void)out_size; (void)ws_size;
  constexpr int B = 4, S = 4096, D = 2048;
  constexpr int M = B * S;  // 16384

  const float* q  = (const float*)d_in[0];
  const float* Wq = (const float*)d_in[1];
  const float* Wk = (const float*)d_in[2];
  const float* Wv = (const float*)d_in[3];
  const float* Wo = (const float*)d_in[4];
  float* out = (float*)d_out;

  uint8_t* w = (uint8_t*)d_ws;
  const size_t szProj = (size_t)M * D * 2;        // 64 MB each (bf16)
  const size_t szP    = (size_t)B * S * S * 2;    // 128 MB (bf16 scores)
  unsigned short* Qb = (unsigned short*)(w);
  unsigned short* Kb = (unsigned short*)(w + szProj);
  unsigned short* Vb = (unsigned short*)(w + 2 * szProj);
  unsigned short* Ob = (unsigned short*)(w + 3 * szProj);
  unsigned short* Pb = (unsigned short*)(w + 4 * szProj);
  float*          rs = (float*)(w + 4 * szProj + szP);

  const dim3 blk(256);
  const float scale = 0.022097086912079608f;  // 1/sqrt(2048)

  // 1) Projections: Qb/Kb/Vb = q @ W (fp32 -> bf16, WMMA)
  {
    dim3 g(D / 128, M / 128, 1);
    wmma_gemm<0, 0, 0><<<g, blk, 0, stream>>>(q, Wq, Qb, M, D, D, D, D, D,
                                              0, 0, 0, 1.f, nullptr, 0);
    wmma_gemm<0, 0, 0><<<g, blk, 0, stream>>>(q, Wk, Kb, M, D, D, D, D, D,
                                              0, 0, 0, 1.f, nullptr, 0);
    wmma_gemm<0, 0, 0><<<g, blk, 0, stream>>>(q, Wv, Vb, M, D, D, D, D, D,
                                              0, 0, 0, 1.f, nullptr, 0);
  }

  // 2) RoPE in place on Qb, Kb
  {
    size_t total = (size_t)B * S * (D / 2);
    rope_kernel<<<dim3((unsigned)(total / 256)), blk, 0, stream>>>(Qb, Kb, S, D);
  }

  // 3) Scores: Pb = mask(scale * Qb @ Kb^T) per batch (bf16)
  {
    dim3 g(S / 128, S / 128, B);
    wmma_gemm<1, 2, 1><<<g, blk, 0, stream>>>(Qb, Kb, Pb, S, S, D, D, D, S,
                                              (size_t)S * D, (size_t)S * D,
                                              (size_t)S * S, scale, nullptr, 0);
  }

  // 4) Row softmax in place, record row sums
  softmax_rows<<<dim3(S, B), blk, 0, stream>>>(Pb, rs, S);

  // 5) Ob = (Pb @ Vb) / rowsum  (causal K-clip)
  {
    dim3 g(D / 128, S / 128, B);
    wmma_gemm<1, 1, 2><<<g, blk, 0, stream>>>(Pb, Vb, Ob, S, D, S, S, D, D,
                                              (size_t)S * S, (size_t)S * D,
                                              (size_t)S * D, 1.f, rs, (size_t)S);
  }

  // 6) out = Ob @ Wo (fp32 output)
  {
    dim3 g(D / 128, M / 128, 1);
    wmma_gemm<1, 0, 3><<<g, blk, 0, stream>>>(Ob, Wo, out, M, D, D, D, D, D,
                                              0, 0, 0, 1.f, nullptr, 0);
  }
}
